// MultimodalUNet_62251255988274
// MI455X (gfx1250) — matrix operand order repaired and checked by reading, not compiled, not run on hardware
//
#include <hip/hip_runtime.h>

typedef __attribute__((ext_vector_type(16))) _Float16 v16h;
typedef __attribute__((ext_vector_type(8)))  _Float16 v8h;
typedef __attribute__((ext_vector_type(4)))  _Float16 v4h;
typedef __attribute__((ext_vector_type(8)))  float    v8f;
typedef __attribute__((ext_vector_type(4)))  float    v4f;

#define VLEN 16384
#define TTOT 20480
#define SCALE 0.35355339059327379f   // 64^-0.25

#define WMMA_F32_F16(a, b, c) \
  __builtin_amdgcn_wmma_f32_16x16x32_f16(false, (a), false, (b), (short)0, (c), false, false)

// ---- wide WMMA fragment loaders (CDNA5 ISA 7.12.2, wave32) ----
// A 16x32 f16 from row-major src: element (m,k) = p[m*ld + k].
// Lane's elements = one contiguous 16-half run (k = 16h..16h+15) -> 2x 16B loads.
__device__ __forceinline__ v16h load_a_row(const _Float16* p, int ld) {
  int lane = threadIdx.x & 31;
  int m = lane & 15, h = lane >> 4;
  const _Float16* q = p + (long)m * ld + 16 * h;
  v8h lo = *(const v8h*)q;          // k = 16h + 0..7   -> elements 0..7
  v8h hi = *(const v8h*)(q + 8);    // k = 16h + 8..15  -> elements 8..15
  v16h r;
#pragma unroll
  for (int e = 0; e < 8; ++e) { r[e] = lo[e]; r[e + 8] = hi[e]; }
  return r;
}
// B 32x16 f16 with K contiguous per column: element (k,n) = p[n*ld + k].
// Lane n reads k = 16h .. 16h+15 -> one 32B run -> 2x 16B loads.
__device__ __forceinline__ v16h load_b_kc(const _Float16* p, long ld) {
  int lane = threadIdx.x & 31;
  int n = lane & 15, h = lane >> 4;
  const _Float16* q = p + (long)n * ld + 16 * h;
  v8h lo = *(const v8h*)q;
  v8h hi = *(const v8h*)(q + 8);
  v16h r;
#pragma unroll
  for (int e = 0; e < 8; ++e) { r[e] = lo[e]; r[e + 8] = hi[e]; }
  return r;
}
// C/D 16x16 f32: element r -> (m = r + 8*(lane>>4), n = lane&15)

// ===== K1: f16 convert + transpose to [t][c] + GN stats (LDS 32x32 tile) =====
__global__ void k_prep(const float* __restrict__ video, const float* __restrict__ audio,
                       _Float16* __restrict__ xv16, _Float16* __restrict__ xa16,
                       float* __restrict__ stats) {
  __shared__ _Float16 tile[32][33];
  int tid = threadIdx.x, bid = blockIdx.x;
  const float* src; _Float16* dst; int which, b; long sstride;
  if (bid < 32) {                        // video: b(2) x f(16); src [c][p] stride 1024
    which = 0; b = bid >> 4; int f = bid & 15;
    src = video + (long)(b * 16 + f) * 256 * 1024;
    dst = xv16 + ((long)b * 16384 + f * 1024) * 256;   // [t][c]
    sstride = 1024;
  } else {                               // audio: b(2) x seg(4); src [c][l] stride 4096
    which = 1; int id = bid - 32; b = id >> 2; int seg = id & 3;
    src = audio + (long)b * 256 * 4096 + seg * 1024;
    dst = xa16 + ((long)b * 4096 + seg * 1024) * 256;
    sstride = 4096;
  }
  int cc = tid >> 3;            // 0..31 tile row (channel)
  int p4 = (tid & 7) * 4;       // 0..28 tile col base (position)
  for (int c0 = 0; c0 < 256; c0 += 32) {
    float s = 0.f, s2 = 0.f;
    for (int p0 = 0; p0 < 1024; p0 += 32) {
      v4f x = *(const v4f*)(src + (long)(c0 + cc) * sstride + p0 + p4);
      s  += x[0] + x[1] + x[2] + x[3];
      s2 += x[0]*x[0] + x[1]*x[1] + x[2]*x[2] + x[3]*x[3];
      tile[cc][p4 + 0] = (_Float16)x[0];
      tile[cc][p4 + 1] = (_Float16)x[1];
      tile[cc][p4 + 2] = (_Float16)x[2];
      tile[cc][p4 + 3] = (_Float16)x[3];
      __syncthreads();
      int rw = tid >> 3, c4 = (tid & 7) * 4;          // write [t][c], c contiguous
      v4h o; o[0] = tile[c4][rw]; o[1] = tile[c4 + 1][rw];
      o[2] = tile[c4 + 2][rw];   o[3] = tile[c4 + 3][rw];
      *(v4h*)(dst + (long)(p0 + rw) * 256 + c0 + c4) = o;
      __syncthreads();
    }
    // reduce over the 8 threads sharing channel c0+cc (consecutive lanes)
    s  += __shfl_xor(s, 1, 32);  s  += __shfl_xor(s, 2, 32);  s  += __shfl_xor(s, 4, 32);
    s2 += __shfl_xor(s2, 1, 32); s2 += __shfl_xor(s2, 2, 32); s2 += __shfl_xor(s2, 4, 32);
    if ((tid & 7) == 0) {
      int g = (c0 + cc) >> 3;
      atomicAdd(&stats[(which * 2 + b) * 64 + g * 2 + 0], s);
      atomicAdd(&stats[(which * 2 + b) * 64 + g * 2 + 1], s2);
    }
  }
}

// ===== K2: per-channel affine from group stats =====
__global__ void k_affine(const float* __restrict__ gvs, const float* __restrict__ gvb,
                         const float* __restrict__ gas, const float* __restrict__ gab,
                         const float* __restrict__ stats,
                         float* __restrict__ sArr, float* __restrict__ tArr) {
  int i = threadIdx.x;                 // 1024 = which(2) x b(2) x c(256)
  int which = i >> 9, b = (i >> 8) & 1, c = i & 255, g = c >> 3;
  float n = which ? (8.f * 4096.f) : (8.f * 16384.f);
  float sum = stats[(which * 2 + b) * 64 + g * 2];
  float ss  = stats[(which * 2 + b) * 64 + g * 2 + 1];
  float mu = sum / n;
  float var = ss / n - mu * mu;
  float r = rsqrtf(var + 1e-5f);
  float sc = (which ? gas[c] : gvs[c]) * r;
  sArr[i] = sc;
  tArr[i] = (which ? gab[c] : gvb[c]) - mu * sc;
}

// ===== K3: fold GN affine into QKV weights; convert proj weights =====
__global__ void k_fold(const float* __restrict__ w_vqkv, const float* __restrict__ b_vqkv,
                       const float* __restrict__ w_aqkv, const float* __restrict__ b_aqkv,
                       const float* __restrict__ w_vproj, const float* __restrict__ w_aproj,
                       const float* __restrict__ sArr, const float* __restrict__ tArr,
                       _Float16* __restrict__ weff16, float* __restrict__ beff,
                       _Float16* __restrict__ wproj16) {
  __shared__ float red[256];
  int tid = threadIdx.x, bid = blockIdx.x;
  if (bid < 3072) {
    int which = bid / 1536, b = (bid / 768) & 1, o = bid % 768;
    const float* W = which ? w_aqkv : w_vqkv;
    float wv = W[o * 256 + tid];
    float sc = sArr[which * 512 + b * 256 + tid];
    float tb = tArr[which * 512 + b * 256 + tid];
    weff16[((long)(which * 2 + b) * 768 + o) * 256 + tid] = (_Float16)(wv * sc);
    red[tid] = wv * tb;
    __syncthreads();
    for (int w = 128; w > 0; w >>= 1) { if (tid < w) red[tid] += red[tid + w]; __syncthreads(); }
    if (tid == 0) {
      const float* bb = which ? b_aqkv : b_vqkv;
      beff[(which * 2 + b) * 768 + o] = bb[o] + red[0];
    }
  } else {
    int j = bid - 3072;
    int which = j >> 8, o = j & 255;
    const float* W = which ? w_aproj : w_vproj;
    wproj16[which * 65536 + o * 256 + tid] = (_Float16)W[o * 256 + tid];
  }
}

// ===== K4: QKV GEMM (WMMA, 16x64 per wave); q/k scaled at store =====
// q,k -> [bh][t][64]; v -> [bh][64][TTOT]
__global__ void k_qkv(const _Float16* __restrict__ weff16, const float* __restrict__ beff,
                      const _Float16* __restrict__ xv16, const _Float16* __restrict__ xa16,
                      _Float16* __restrict__ q16, _Float16* __restrict__ k16,
                      _Float16* __restrict__ v16) {
  int wave = blockIdx.x * 8 + (threadIdx.x >> 5);   // 30720 waves exact
  int b = wave / 15360;
  int r = wave % 15360;
  const _Float16* A; const _Float16* X; int ot, biasbase, tglob0;
  if (r < 12288) {                                   // video: 48 o-tiles x 256 t-groups
    ot = r >> 8; int tg = r & 255;
    A = weff16 + ((long)b * 768 + ot * 16) * 256;
    X = xv16 + ((long)b * 16384 + tg * 64) * 256;
    biasbase = b * 768; tglob0 = tg * 64;
  } else {                                           // audio: 48 x 64
    int ra = r - 12288;
    ot = ra >> 6; int tg = ra & 63;
    A = weff16 + ((long)(2 + b) * 768 + ot * 16) * 256;
    X = xa16 + ((long)b * 4096 + tg * 64) * 256;
    biasbase = (2 + b) * 768; tglob0 = VLEN + tg * 64;
  }
  v8f acc[4] = {{}, {}, {}, {}};
#pragma unroll
  for (int c0 = 0; c0 < 256; c0 += 32) {
    v16h a = load_a_row(A + c0, 256);
#pragma unroll
    for (int j = 0; j < 4; ++j) {
      v16h bf = load_b_kc(X + (long)(j * 16) * 256 + c0, 256);
      acc[j] = WMMA_F32_F16(a, bf, acc[j]);
    }
  }
  int lane = threadIdx.x & 31, n = lane & 15, h = lane >> 4;
#pragma unroll
  for (int j = 0; j < 4; ++j) {
    int tglob = tglob0 + j * 16 + n;
#pragma unroll
    for (int e = 0; e < 8; ++e) {
      int o = ot * 16 + e + 8 * h;
      float val = acc[j][e] + beff[biasbase + o];
      if (o < 256) {
        q16[((long)(b * 4 + (o >> 6)) * TTOT + tglob) * 64 + (o & 63)] = (_Float16)(val * SCALE);
      } else if (o < 512) {
        int ko = o - 256;
        k16[((long)(b * 4 + (ko >> 6)) * TTOT + tglob) * 64 + (ko & 63)] = (_Float16)(val * SCALE);
      } else {
        int vo = o - 512;
        v16[((long)(b * 4 + (vo >> 6)) * 64 + (vo & 63)) * TTOT + tglob] = (_Float16)val;
      }
    }
  }
}

// ===== K5: flash attention, one wave = 16 query rows =====
__global__ void k_attn(const _Float16* __restrict__ q16, const _Float16* __restrict__ k16,
                       const _Float16* __restrict__ v16, _Float16* __restrict__ o16) {
  __shared__ __align__(16) _Float16 pbuf[8][16][32];
  int w = threadIdx.x >> 5, lane = threadIdx.x & 31;
  int wt = blockIdx.x * 8 + w;                       // 10240 waves exact
  int bh, f, rt, qt0, kt0, cols;
  if (wt < 8192) {                                   // video q -> audio kv
    rt = wt & 63; f = (wt >> 6) & 15; bh = wt >> 10;
    qt0 = f * 1024 + rt * 16; kt0 = VLEN + f * 256; cols = 256;
  } else {                                           // audio q -> video kv
    int a = wt - 8192;
    rt = a & 15; f = (a >> 4) & 15; bh = a >> 8;
    qt0 = VLEN + f * 256 + rt * 16; kt0 = f * 1024; cols = 1024;
  }
  int b = bh >> 2, hc = (bh & 3) * 64;
  const _Float16* qp = q16 + ((long)bh * TTOT + qt0) * 64;   // [t][64]
  const _Float16* kp = k16 + ((long)bh * TTOT + kt0) * 64;   // [s][64]
  const _Float16* vp = v16 + (long)bh * 64 * TTOT + kt0;     // [c][s]
  _Float16* op = o16 + ((long)b * TTOT + qt0) * 256 + hc;    // [t][256]

  v16h aq0 = load_a_row(qp, 64);        // Q^T tile, c 0..31
  v16h aq1 = load_a_row(qp + 32, 64);   // c 32..63

  v8f o0 = {}, o1 = {}, o2 = {}, o3 = {};
  float rmax[8], rsum[8];
#pragma unroll
  for (int e = 0; e < 8; ++e) { rmax[e] = -3.0e38f; rsum[e] = 0.f; }

  for (int sc = 0; sc < cols; sc += 32) {
    v8f s0 = {}, s1 = {};
    v16h bk = load_b_kc(kp + (long)sc * 64, 64);
    s0 = WMMA_F32_F16(aq0, bk, s0);
    bk = load_b_kc(kp + (long)sc * 64 + 32, 64);
    s0 = WMMA_F32_F16(aq1, bk, s0);
    bk = load_b_kc(kp + (long)(sc + 16) * 64, 64);
    s1 = WMMA_F32_F16(aq0, bk, s1);
    bk = load_b_kc(kp + (long)(sc + 16) * 64 + 32, 64);
    s1 = WMMA_F32_F16(aq1, bk, s1);

    float nm[8], al[8];
#pragma unroll
    for (int e = 0; e < 8; ++e) {                    // row max across 16-lane half
      float m = fmaxf(s0[e], s1[e]);
      m = fmaxf(m, __shfl_xor(m, 1, 32));
      m = fmaxf(m, __shfl_xor(m, 2, 32));
      m = fmaxf(m, __shfl_xor(m, 4, 32));
      m = fmaxf(m, __shfl_xor(m, 8, 32));
      nm[e] = fmaxf(rmax[e], m);
      al[e] = __expf(rmax[e] - nm[e]);
      rmax[e] = nm[e];
    }
#pragma unroll
    for (int e = 0; e < 8; ++e) {
      float p0 = __expf(s0[e] - nm[e]);
      float p1 = __expf(s1[e] - nm[e]);
      float ps = p0 + p1;
      ps += __shfl_xor(ps, 1, 32);
      ps += __shfl_xor(ps, 2, 32);
      ps += __shfl_xor(ps, 4, 32);
      ps += __shfl_xor(ps, 8, 32);
      rsum[e] = rsum[e] * al[e] + ps;
      o0[e] *= al[e]; o1[e] *= al[e]; o2[e] *= al[e]; o3[e] *= al[e];
      int m = e + 8 * (lane >> 4), nn = lane & 15;
      pbuf[w][m][nn] = (_Float16)p0;                 // stage P for A-frag reload
      pbuf[w][m][nn + 16] = (_Float16)p1;
    }
    asm volatile("s_wait_dscnt 0" ::: "memory");     // intra-wave LDS RAW
    v16h ap = load_a_row(&pbuf[w][0][0], 32);
    v16h bv = load_b_kc(vp + sc, TTOT);
    o0 = WMMA_F32_F16(ap, bv, o0);
    bv = load_b_kc(vp + sc + (long)16 * TTOT, TTOT);
    o1 = WMMA_F32_F16(ap, bv, o1);
    bv = load_b_kc(vp + sc + (long)32 * TTOT, TTOT);
    o2 = WMMA_F32_F16(ap, bv, o2);
    bv = load_b_kc(vp + sc + (long)48 * TTOT, TTOT);
    o3 = WMMA_F32_F16(ap, bv, o3);
  }
  int n = lane & 15, h = lane >> 4;
#pragma unroll
  for (int e = 0; e < 8; ++e) {
    float inv = 1.0f / rsum[e];
    int m = e + 8 * h;
    op[(long)m * 256 + n]      = (_Float16)(o0[e] * inv);
    op[(long)m * 256 + n + 16] = (_Float16)(o1[e] * inv);
    op[(long)m * 256 + n + 32] = (_Float16)(o2[e] * inv);
    op[(long)m * 256 + n + 48] = (_Float16)(o3[e] * inv);
  }
}

// ===== K6: projection GEMMs (16x64 per wave) + bias + residual =====
__global__ void k_proj(const _Float16* __restrict__ wproj16, const _Float16* __restrict__ o16,
                       const float* __restrict__ b_vproj, const float* __restrict__ b_aproj,
                       const float* __restrict__ video, const float* __restrict__ audio,
                       float* __restrict__ out) {
  int wave = blockIdx.x * 8 + (threadIdx.x >> 5);    // 10240 waves exact
  int lane = threadIdx.x & 31, n = lane & 15, h = lane >> 4;
  if (wave < 8192) {                                 // video: b(2) x ot(16) x tg(256)
    int tg = wave & 255, ot = (wave >> 8) & 15, b = wave >> 12;
    const _Float16* A = wproj16 + ot * 16 * 256;
    const _Float16* Bb = o16 + ((long)b * TTOT + tg * 64) * 256;
    v8f acc[4] = {{}, {}, {}, {}};
#pragma unroll
    for (int c0 = 0; c0 < 256; c0 += 32) {
      v16h a = load_a_row(A + c0, 256);
#pragma unroll
      for (int j = 0; j < 4; ++j) {
        v16h bf = load_b_kc(Bb + (long)(j * 16) * 256 + c0, 256);
        acc[j] = WMMA_F32_F16(a, bf, acc[j]);
      }
    }
#pragma unroll
    for (int j = 0; j < 4; ++j) {
      int t = tg * 64 + j * 16 + n, f = t >> 10, p = t & 1023;
#pragma unroll
      for (int e = 0; e < 8; ++e) {
        int o = ot * 16 + e + 8 * h;
        long idx = ((long)(b * 16 + f) * 256 + o) * 1024 + p;
        out[idx] = acc[j][e] + b_vproj[o] + video[idx];
      }
    }
  } else {                                           // audio: b(2) x ot(16) x tg(64)
    int wa = wave - 8192;
    int tg = wa & 63, ot = (wa >> 6) & 15, b = wa >> 10;
    const _Float16* A = wproj16 + 65536 + ot * 16 * 256;
    const _Float16* Bb = o16 + ((long)b * TTOT + VLEN + tg * 64) * 256;
    v8f acc[4] = {{}, {}, {}, {}};
#pragma unroll
    for (int c0 = 0; c0 < 256; c0 += 32) {
      v16h a = load_a_row(A + c0, 256);
#pragma unroll
      for (int j = 0; j < 4; ++j) {
        v16h bf = load_b_kc(Bb + (long)(j * 16) * 256 + c0, 256);
        acc[j] = WMMA_F32_F16(a, bf, acc[j]);
      }
    }
#pragma unroll
    for (int j = 0; j < 4; ++j) {
      int l = tg * 64 + j * 16 + n;
#pragma unroll
      for (int e = 0; e < 8; ++e) {
        int o = ot * 16 + e + 8 * h;
        long aidx = ((long)b * 256 + o) * 4096 + l;
        out[8388608 + aidx] = acc[j][e] + b_aproj[o] + audio[aidx];
      }
    }
  }
}

extern "C" void kernel_launch(void* const* d_in, const int* in_sizes, int n_in,
                              void* d_out, int out_size, void* d_ws, size_t ws_size,
                              hipStream_t stream) {
  (void)in_sizes; (void)n_in; (void)out_size; (void)ws_size;
  const float* video   = (const float*)d_in[0];
  const float* audio   = (const float*)d_in[1];
  const float* gvs     = (const float*)d_in[2];
  const float* gvb     = (const float*)d_in[3];
  const float* gas     = (const float*)d_in[4];
  const float* gab     = (const float*)d_in[5];
  const float* w_vqkv  = (const float*)d_in[6];
  const float* b_vqkv  = (const float*)d_in[7];
  const float* w_aqkv  = (const float*)d_in[8];
  const float* b_aqkv  = (const float*)d_in[9];
  const float* w_vproj = (const float*)d_in[10];
  const float* b_vproj = (const float*)d_in[11];
  const float* w_aproj = (const float*)d_in[12];
  const float* b_aproj = (const float*)d_in[13];
  float* out = (float*)d_out;

  char* ws = (char*)d_ws;
  size_t off = 0;
  auto carve = [&](size_t bytes) -> char* {
    char* p = ws + off;
    off = (off + bytes + 255) & ~(size_t)255;
    return p;
  };
  _Float16* xv16    = (_Float16*)carve(2UL * 16384 * 256 * 2);   // [b][t][c]
  _Float16* xa16    = (_Float16*)carve(2UL * 4096 * 256 * 2);    // [b][l][c]
  _Float16* q16     = (_Float16*)carve(8UL * 20480 * 64 * 2);    // [bh][t][ch]
  _Float16* k16     = (_Float16*)carve(8UL * 20480 * 64 * 2);    // [bh][s][ch]
  _Float16* v16     = (_Float16*)carve(8UL * 64 * 20480 * 2);    // [bh][ch][s]
  _Float16* o16     = (_Float16*)carve(2UL * 20480 * 256 * 2);   // [b][t][c]
  _Float16* weff16  = (_Float16*)carve(4UL * 768 * 256 * 2);
  _Float16* wproj16 = (_Float16*)carve(2UL * 256 * 256 * 2);
  float*    beff    = (float*)carve(4UL * 768 * 4);
  float*    stats   = (float*)carve(256 * 4);
  float*    sArr    = (float*)carve(1024 * 4);
  float*    tArr    = (float*)carve(1024 * 4);

  hipMemsetAsync(stats, 0, 256 * 4, stream);
  k_prep<<<40, 256, 0, stream>>>(video, audio, xv16, xa16, stats);
  k_affine<<<1, 1024, 0, stream>>>(gvs, gvb, gas, gab, stats, sArr, tArr);
  k_fold<<<3584, 256, 0, stream>>>(w_vqkv, b_vqkv, w_aqkv, b_aqkv, w_vproj, w_aproj,
                                   sArr, tArr, weff16, beff, wproj16);
  k_qkv<<<3840, 256, 0, stream>>>(weff16, beff, xv16, xa16, q16, k16, v16);
  k_attn<<<1280, 256, 0, stream>>>(q16, k16, v16, o16);
  k_proj<<<1280, 256, 0, stream>>>(wproj16, o16, b_vproj, b_aproj, video, audio, out);
}
